// QuadrantAttention_42245298323781
// MI455X (gfx1250) — matrix operand order repaired and checked by reading, not compiled, not run on hardware
//
#include <hip/hip_runtime.h>
#include <hip/hip_bf16.h>
#include <math.h>

// ---------------------------------------------------------------------------
// QuadrantAttention for MI455X (gfx1250): fold quadrant mix into Wk, then
// standard MHA: 4x f16 WMMA GEMMs + f16 WMMA flash attention.
// CDNA5 paths: v_wmma_f32_16x16x32_f16, global_load_async_to_lds_b128
// (ASYNCcnt) for copy-through staging, global_prefetch_b8 for next tiles.
// ---------------------------------------------------------------------------

typedef __attribute__((ext_vector_type(16))) _Float16 v16h;
typedef __attribute__((ext_vector_type(8)))  _Float16 v8h;
typedef __attribute__((ext_vector_type(8)))  float    v8f;

#define WMMA_F16(a, b, c) \
  __builtin_amdgcn_wmma_f32_16x16x32_f16(false, (a), false, (b), (short)0, (c), false, false)

static __device__ __forceinline__ int lane_id() { return (int)(threadIdx.x & 31); }

// Async copy 16 bytes global -> LDS (per lane). LDS offset = low 32 bits of
// the flat shared-aperture pointer.
static __device__ __forceinline__ void async_copy_b128(const void* gptr, void* lptr) {
  unsigned long long ga = (unsigned long long)gptr;
  unsigned la = (unsigned)(unsigned long long)lptr;
  asm volatile("global_load_async_to_lds_b128 %0, %1, off" :: "v"(la), "v"(ga) : "memory");
}
static __device__ __forceinline__ void wait_async0() {
  asm volatile("s_wait_asynccnt 0x0" ::: "memory");
}

// A fragment: 16x32 f16 tile at lds[row0..row0+15][k0..k0+31], row stride ld.
// Lanes 0-15 (row M=lane) hold K 0..15; lanes 16-31
// hold K 16..31.
static __device__ __forceinline__ v16h load_a_frag(const _Float16* lds, int row0, int k0, int ld) {
  const int lane = lane_id();
  const _Float16* p = lds + (size_t)(row0 + (lane & 15)) * ld + k0;
  const int kb = (lane >> 4) << 4; // 0 or 16
  v16h a;
#pragma unroll
  for (int i = 0; i < 16; ++i) a[i] = p[kb + i];
  return a;
}

// B fragment: 32x16 f16 read from K-transposed LDS storage ldsT[n][k]
// (column n = lane&15; lanes 0-15 hold K 0..15, lanes 16-31 hold K 16..31).
static __device__ __forceinline__ v16h load_b_frag(const _Float16* ldsT, int n0, int k0, int ld) {
  const int lane = lane_id();
  const _Float16* p = ldsT + (size_t)(n0 + (lane & 15)) * ld + k0 + ((lane >> 4) << 4);
  v16h b;
#pragma unroll
  for (int i = 0; i < 16; ++i) b[i] = p[i];
  return b;
}

// ---------------------------------------------------------------------------
// Elementwise f32 -> f16
// ---------------------------------------------------------------------------
__global__ void f32_to_f16_kernel(const float* __restrict__ in, _Float16* __restrict__ out, int n) {
  int i = blockIdx.x * blockDim.x + threadIdx.x;
  if (i < n) out[i] = (_Float16)in[i];
}

// ---------------------------------------------------------------------------
// Build Wk_eff (f16) = per-head quadrant mix of Wk columns, w = softmax(qw).
// ---------------------------------------------------------------------------
__global__ void mix_wk_kernel(const float* __restrict__ Wk, const float* __restrict__ qw,
                              _Float16* __restrict__ out, int Ddim) {
  int i = blockIdx.x * blockDim.x + threadIdx.x;
  if (i >= Ddim * Ddim) return;
  float q0 = qw[0], q1 = qw[1], q2 = qw[2], q3 = qw[3];
  float mx = fmaxf(fmaxf(q0, q1), fmaxf(q2, q3));
  float e0 = __expf(q0 - mx), e1 = __expf(q1 - mx), e2 = __expf(q2 - mx), e3 = __expf(q3 - mx);
  float inv = 1.0f / (e0 + e1 + e2 + e3);
  float w0 = e0 * inv, w1 = e1 * inv, w2 = e2 * inv, w3 = e3 * inv;

  int r = i / Ddim, c = i - r * Ddim;
  int cin = c & 63;        // channel within head
  int hb  = c & ~63;       // head column base
  const float* row = Wk + (size_t)r * Ddim + hb;
  float v;
  if (cin < 32) v = w0 * row[cin]       + w1 * row[cin + 32];
  else          v = w2 * row[cin - 32]  + w3 * row[cin];
  out[i] = (_Float16)v;
}

// ---------------------------------------------------------------------------
// Tiled WMMA GEMM: C[M,N] = A[M,K] * B[K,N], f16 in, f32 accumulate.
// Block = 128x128, 4 waves (2x2), wave tile = 64x64 (4x4 WMMA tiles), BK=32.
// A tile staged via async global->LDS b128; B tile staged transposed [n][k].
// ---------------------------------------------------------------------------
template <typename OutT>
__global__ __launch_bounds__(128) void gemm_f16_kernel(
    const _Float16* __restrict__ A, const _Float16* __restrict__ Bm,
    OutT* __restrict__ C, int Mdim, int Ndim, int Kdim) {
  __shared__ alignas(16) _Float16 As[128 * 32];   // [m][k]
  __shared__ alignas(16) _Float16 Bs[128 * 32];   // transposed: [n][k]

  const int tid  = threadIdx.x;
  const int warp = tid >> 5;
  const int wr   = warp >> 1;    // 0..1
  const int wc   = warp & 1;     // 0..1
  const int bm   = blockIdx.y * 128;
  const int bn   = blockIdx.x * 128;

  v8f acc[4][4];
#pragma unroll
  for (int i = 0; i < 4; ++i)
#pragma unroll
    for (int j = 0; j < 4; ++j) acc[i][j] = (v8f){};

  for (int kt = 0; kt < Kdim; kt += 32) {
    __syncthreads();
    // Stage A tile 128x32 via async copies (512 chunks of 8 halfs).
#pragma unroll
    for (int it = 0; it < 4; ++it) {
      int idx = it * 128 + tid;
      int r = idx >> 2, cc = (idx & 3) * 8;
      async_copy_b128(&A[(size_t)(bm + r) * Kdim + kt + cc], &As[r * 32 + cc]);
    }
    // Stage B tile 32x128, transposed into Bs[n][k].
#pragma unroll
    for (int it = 0; it < 4; ++it) {
      int idx = it * 128 + tid;
      int r = idx >> 4, cc = (idx & 15) * 8;
      v8h t = *(const v8h*)&Bm[(size_t)(kt + r) * Ndim + bn + cc];
#pragma unroll
      for (int e = 0; e < 8; ++e) Bs[(cc + e) * 32 + r] = t[e];
    }
    // Prefetch next k-tile into L2 while this one computes.
    if (kt + 32 < Kdim) {
      __builtin_prefetch(&A[(size_t)(bm + tid) * Kdim + kt + 32], 0, 0);
      __builtin_prefetch(&Bm[(size_t)(kt + 32 + (tid >> 2)) * Ndim + bn + (tid & 3) * 32], 0, 0);
    }
    wait_async0();
    __syncthreads();

    v16h bfr[4];
#pragma unroll
    for (int ns = 0; ns < 4; ++ns)
      bfr[ns] = load_b_frag(Bs, wc * 64 + ns * 16, 0, 32);
#pragma unroll
    for (int ms = 0; ms < 4; ++ms) {
      v16h a = load_a_frag(As, wr * 64 + ms * 16, 0, 32);
#pragma unroll
      for (int ns = 0; ns < 4; ++ns)
        acc[ms][ns] = WMMA_F16(a, bfr[ns], acc[ms][ns]);
    }
  }

  // C layout: lane<16 -> col=lane, row=g; lane>=16 -> row=8+g.
  const int lane = lane_id();
  const int cn = lane & 15;
  const int rb = (lane >> 4) * 8;
#pragma unroll
  for (int ms = 0; ms < 4; ++ms)
#pragma unroll
    for (int ns = 0; ns < 4; ++ns) {
      int gcol = bn + wc * 64 + ns * 16 + cn;
#pragma unroll
      for (int g = 0; g < 8; ++g) {
        int grow = bm + wr * 64 + ms * 16 + rb + g;
        C[(size_t)grow * Ndim + gcol] = (OutT)acc[ms][ns][g];
      }
    }
}

// ---------------------------------------------------------------------------
// Flash attention: one block per (64-row q-tile, head, batch). 4 waves, each
// owns 16 q rows; K/V processed in 128-wide tiles (32 WMMA per iteration).
// Online softmax in C-register layout, P restaged via per-wave LDS.
// ---------------------------------------------------------------------------
__global__ __launch_bounds__(128) void flash_attn_kernel(
    const _Float16* __restrict__ Q, const _Float16* __restrict__ K,
    const _Float16* __restrict__ V, _Float16* __restrict__ O,
    int Tlen, int nheads) {
  __shared__ alignas(16) _Float16 Qs[64 * 64];       // [q][c]
  __shared__ alignas(16) _Float16 Ks[128 * 64];      // [j][c] == B^T storage for Q.K^T
  __shared__ alignas(16) _Float16 Vt[64 * 128];      // [c][j] == B^T storage for P.V
  __shared__ alignas(16) _Float16 Ps[4][16 * 128];   // per-wave P staging [m][j]

  const int tid  = threadIdx.x;
  const int warp = tid >> 5;
  const int lane = tid & 31;
  const int qt = blockIdx.x, h = blockIdx.y, bb = blockIdx.z;
  const int Dfull = nheads * 64;
  const size_t rowbase = (size_t)bb * Tlen;
  const int colbase = h * 64;
  const float scale = 0.17677669529663687f;   // 1/sqrt(32)

  // Load Q tile (64x64) via async copies.
  const int q0 = qt * 64;
#pragma unroll
  for (int it = 0; it < 4; ++it) {
    int idx = it * 128 + tid;
    int r = idx >> 3, cc = (idx & 7) * 8;
    async_copy_b128(&Q[(rowbase + q0 + r) * Dfull + colbase + cc], &Qs[r * 64 + cc]);
  }
  wait_async0();
  __syncthreads();
  v16h qa0 = load_a_frag(Qs, warp * 16, 0, 64);
  v16h qa1 = load_a_frag(Qs, warp * 16, 32, 64);
  // Fold the 1/sqrt(h) scale into the Q fragments once.
#pragma unroll
  for (int i = 0; i < 16; ++i) {
    qa0[i] = (_Float16)((float)qa0[i] * scale);
    qa1[i] = (_Float16)((float)qa1[i] * scale);
  }

  v8f o[4];
#pragma unroll
  for (int u = 0; u < 4; ++u) o[u] = (v8f){};
  float mrow[8], lrow[8];
#pragma unroll
  for (int g = 0; g < 8; ++g) { mrow[g] = -1e30f; lrow[g] = 0.0f; }

  for (int kt = 0; kt < Tlen; kt += 128) {
    __syncthreads();
    // Stage K tile 128x64 [j][c] (async copy-through) and V transposed [c][j].
#pragma unroll
    for (int it = 0; it < 8; ++it) {
      int idx = it * 128 + tid;
      int r = idx >> 3, cc = (idx & 7) * 8;
      async_copy_b128(&K[(rowbase + kt + r) * Dfull + colbase + cc], &Ks[r * 64 + cc]);
      v8h tv = *(const v8h*)&V[(rowbase + kt + r) * Dfull + colbase + cc];
#pragma unroll
      for (int e = 0; e < 8; ++e) Vt[(cc + e) * 128 + r] = tv[e];
    }
    if (kt + 128 < Tlen)
      __builtin_prefetch(&K[(rowbase + kt + 128 + tid) * Dfull + colbase], 0, 0);
    wait_async0();
    __syncthreads();

    // S(16x128) = Qhat(16x64) . K^T : 8 tiles x 2 k-chunks = 16 WMMA.
    v8f s[8];
#pragma unroll
    for (int t = 0; t < 8; ++t) {
      v8f sa = (v8f){};
      v16h b0 = load_b_frag(Ks, t * 16, 0, 64);
      sa = WMMA_F16(qa0, b0, sa);
      v16h b1 = load_b_frag(Ks, t * 16, 32, 64);
      sa = WMMA_F16(qa1, b1, sa);
      s[t] = sa;
    }

    // Online softmax (per row g, replicated over 16-lane halves).
    float alpha[8];
#pragma unroll
    for (int g = 0; g < 8; ++g) {
      float mx = s[0][g];
#pragma unroll
      for (int t = 1; t < 8; ++t) mx = fmaxf(mx, s[t][g]);
#pragma unroll
      for (int off = 1; off < 16; off <<= 1)
        mx = fmaxf(mx, __shfl_xor(mx, off, 32));
      float mnew = fmaxf(mrow[g], mx);
      alpha[g] = __expf(mrow[g] - mnew);
      mrow[g] = mnew;
      float rs = 0.0f;
#pragma unroll
      for (int t = 0; t < 8; ++t) {
        float p = __expf(s[t][g] - mnew);
        s[t][g] = p;
        rs += p;
      }
#pragma unroll
      for (int off = 1; off < 16; off <<= 1)
        rs += __shfl_xor(rs, off, 32);
      lrow[g] = lrow[g] * alpha[g] + rs;
#pragma unroll
      for (int u = 0; u < 4; ++u) o[u][g] *= alpha[g];
    }

    // Re-stage P (f16) into A-fragment-friendly [m][j] layout.
    _Float16* pw = Ps[warp];
    const int cn = lane & 15;
    const int rb = (lane >> 4) * 8;
#pragma unroll
    for (int t = 0; t < 8; ++t)
#pragma unroll
      for (int g = 0; g < 8; ++g)
        pw[(rb + g) * 128 + t * 16 + cn] = (_Float16)s[t][g];
    __syncthreads();

    // O(16x64) += P(16x128) . V(128x64) : 4 d-tiles x 4 k-chunks = 16 WMMA.
    v16h pa[4];
#pragma unroll
    for (int jc = 0; jc < 4; ++jc) pa[jc] = load_a_frag(pw, 0, jc * 32, 128);
#pragma unroll
    for (int u = 0; u < 4; ++u)
#pragma unroll
      for (int jc = 0; jc < 4; ++jc) {
        v16h b = load_b_frag(Vt, u * 16, jc * 32, 128);
        o[u] = WMMA_F16(pa[jc], b, o[u]);
      }
  }

  // Epilogue: O = acc / l, store f16 into (B*T, D) attention-output matrix.
  const int cn = lane & 15;
  const int rb = (lane >> 4) * 8;
#pragma unroll
  for (int u = 0; u < 4; ++u)
#pragma unroll
    for (int g = 0; g < 8; ++g) {
      float val = o[u][g] / lrow[g];
      int grow = q0 + warp * 16 + rb + g;
      O[(rowbase + grow) * Dfull + colbase + u * 16 + cn] = (_Float16)val;
    }
}

// ---------------------------------------------------------------------------
// Host launcher
// ---------------------------------------------------------------------------
extern "C" void kernel_launch(void* const* d_in, const int* in_sizes, int n_in,
                              void* d_out, int out_size, void* d_ws, size_t ws_size,
                              hipStream_t stream) {
  (void)in_sizes; (void)n_in; (void)out_size; (void)ws_size;
  const float* x  = (const float*)d_in[0];
  const float* Wq = (const float*)d_in[1];
  const float* Wk = (const float*)d_in[2];
  const float* Wv = (const float*)d_in[3];
  const float* Wo = (const float*)d_in[4];
  const float* qw = (const float*)d_in[5];
  float* out = (float*)d_out;

  const int Bc = 2, Tc = 2048, Dc = 1024, Hc = 16;
  const int Mc = Bc * Tc;                 // 4096
  const size_t xd = (size_t)Mc * Dc;      // 4M elems
  const size_t wd = (size_t)Dc * Dc;      // 1M elems

  char* ws = (char*)d_ws;
  _Float16* xh  = (_Float16*)ws;            ws += xd * sizeof(_Float16);
  _Float16* Wqh = (_Float16*)ws;            ws += wd * sizeof(_Float16);
  _Float16* Wkh = (_Float16*)ws;            ws += wd * sizeof(_Float16);
  _Float16* Wvh = (_Float16*)ws;            ws += wd * sizeof(_Float16);
  _Float16* Woh = (_Float16*)ws;            ws += wd * sizeof(_Float16);
  _Float16* Qh  = (_Float16*)ws;            ws += xd * sizeof(_Float16);
  _Float16* Kh  = (_Float16*)ws;            ws += xd * sizeof(_Float16);
  _Float16* Vh  = (_Float16*)ws;            ws += xd * sizeof(_Float16);
  _Float16* Ah  = (_Float16*)ws;            ws += xd * sizeof(_Float16);

  // 1) Precision conversion + Wk quadrant folding.
  f32_to_f16_kernel<<<(int)((xd + 255) / 256), 256, 0, stream>>>(x,  xh,  (int)xd);
  f32_to_f16_kernel<<<(int)((wd + 255) / 256), 256, 0, stream>>>(Wq, Wqh, (int)wd);
  f32_to_f16_kernel<<<(int)((wd + 255) / 256), 256, 0, stream>>>(Wv, Wvh, (int)wd);
  f32_to_f16_kernel<<<(int)((wd + 255) / 256), 256, 0, stream>>>(Wo, Woh, (int)wd);
  mix_wk_kernel<<<(int)((wd + 255) / 256), 256, 0, stream>>>(Wk, qw, Wkh, Dc);

  // 2) Projections Q = x Wq, K_eff = x Wk_eff, V = x Wv.
  dim3 gg(Dc / 128, Mc / 128);
  gemm_f16_kernel<_Float16><<<gg, 128, 0, stream>>>(xh, Wqh, Qh, Mc, Dc, Dc);
  gemm_f16_kernel<_Float16><<<gg, 128, 0, stream>>>(xh, Wkh, Kh, Mc, Dc, Dc);
  gemm_f16_kernel<_Float16><<<gg, 128, 0, stream>>>(xh, Wvh, Vh, Mc, Dc, Dc);

  // 3) Flash attention per (b, h, 64-row q tile).
  flash_attn_kernel<<<dim3(Tc / 64, Hc, Bc), 128, 0, stream>>>(Qh, Kh, Vh, Ah, Tc, Hc);

  // 4) Output projection (f32 result straight into d_out).
  gemm_f16_kernel<float><<<gg, 128, 0, stream>>>(Ah, Woh, out, Mc, Dc, Dc);
}
